// LocalSelfAttentionMultiHeadSum_32959579030108
// MI455X (gfx1250) — matrix operand order repaired and checked by reading, not compiled, not run on hardware
//
#include <hip/hip_runtime.h>
#include <hip/hip_bf16.h>

// ---------------- problem constants ----------------
constexpr int   Bb  = 8;      // batch
constexpr int   Cc  = 512;    // channels
constexpr int   Tt  = 32;     // time
constexpr int   Pp  = 49;     // H*W
constexpr int   P64 = 64;     // padded P
constexpr int   NH  = 8;      // heads
constexpr int   Ee  = 64;     // embed per head
constexpr int   KW  = 7;      // window
constexpr int   Ll  = 343;    // KW*Pp
constexpr int   L352= 352;    // padded L
constexpr float EPS = 1e-5f;

// ---------------- workspace layout (bytes) ----------------
constexpr size_t OFF_MEAN = 0;
constexpr size_t OFF_RSTD = OFF_MEAN + 2048;
constexpr size_t OFF_BOM  = OFF_RSTD + 2048;
constexpr size_t OFF_WQH  = 8192;
constexpr size_t SZ_W     = (size_t)NH * Ee * Cc * 2;           // 512 KB
constexpr size_t OFF_WKH  = OFF_WQH + SZ_W;
constexpr size_t OFF_WVH  = OFF_WKH + SZ_W;
constexpr size_t OFF_WOTH = OFF_WVH + SZ_W;                     // [c][(n,e)]
constexpr size_t OFF_XH   = OFF_WOTH + SZ_W;
constexpr size_t SZ_XH    = (size_t)Bb * Tt * P64 * Cc * 2;     // 16 MB
constexpr size_t OFF_QT   = OFF_XH + SZ_XH;
constexpr size_t SZ_PROJ  = (size_t)Bb * NH * Tt * P64 * Ee * 2;// 16 MB
constexpr size_t OFF_KT   = OFF_QT + SZ_PROJ;
constexpr size_t OFF_VE   = OFF_KT + SZ_PROJ;
constexpr size_t OFF_YT   = OFF_VE + SZ_PROJ;                   // [b][t][p][n][e]

// ---------------- LDS layout for attention kernel ----------------
constexpr size_t SM_K  = (size_t)L352 * 64 * 2;   // k[l][e]      45056 B
constexpr size_t SM_V  = (size_t)64 * L352 * 2;   // v[e][l]      45056 B
constexpr size_t SM_Q  = (size_t)64 * 64 * 2;     // q[p][e]       8192 B
constexpr size_t SM_AF = (size_t)64 * L352 * 4;   // alpha f32    90112 B
constexpr size_t SM_AH = (size_t)64 * L352 * 2;   // alpha bf16   45056 B
constexpr size_t SMEM_ATTN = SM_K + SM_V + SM_Q + SM_AF + SM_AH;  // 233472 B < 320 KB

// ---------------- vector types / helpers ----------------
typedef __attribute__((ext_vector_type(16))) __bf16         v16bf;
typedef __attribute__((ext_vector_type(8)))  float          v8f;
typedef __attribute__((ext_vector_type(8)))  unsigned short v8us;
typedef __attribute__((ext_vector_type(16))) unsigned short v16us;

union frag_u { v16us us; v16bf bf; struct { v8us lo, hi; } h; };

static __device__ __forceinline__ unsigned short f2bf(float f) {
  unsigned u = __float_as_uint(f);
  unsigned r = ((u >> 16) & 1u) + 0x7fffu;   // round to nearest even
  return (unsigned short)((u + r) >> 16);
}

// A fragment: 16x32 bf16, row-major source [M][K], stride in halfwords.
// lane = m + 16*hi ; K chunks {koff..koff+7} and {16+koff..16+koff+7}, koff=8*hi.
static __device__ __forceinline__ v16bf load_fragA(const unsigned short* base, int stride,
                                                   int row0, int k0, int lane) {
  int m    = lane & 15;
  int koff = (lane >> 4) * 8;
  const unsigned short* p = base + (size_t)(row0 + m) * stride + k0 + koff;
  frag_u u;
  u.h.lo = *(const v8us*)p;
  u.h.hi = *(const v8us*)(p + 16);
  return u.bf;
}

// B fragment: 32x16 bf16, source is B^T row-major [N][K].
// lane = n + 16*hi ; K chunks {koff..koff+7} and {16+koff..16+koff+7}, koff=8*hi.
static __device__ __forceinline__ v16bf load_fragB(const unsigned short* base, int stride,
                                                   int col0, int k0, int lane) {
  const unsigned short* p = base + (size_t)(col0 + (lane & 15)) * stride + k0 + ((lane >> 4) * 8);
  frag_u u;
  u.h.lo = *(const v8us*)p;
  u.h.hi = *(const v8us*)(p + 16);
  return u.bf;
}

static __device__ __forceinline__ v8f wmma_bf16(v16bf a, v16bf b, v8f c) {
  return __builtin_amdgcn_wmma_f32_16x16x32_bf16(false, a, false, b, (short)0, c, false, false);
}

// ================= kernel 1: BN stats per channel =================
__global__ __launch_bounds__(256) void bnstats_kernel(const float* __restrict__ x,
                                                      float* mean, float* rstd) {
  __shared__ float ssum[256], ssq[256];
  int c = blockIdx.x;
  float s = 0.f, q = 0.f;
  for (int j = threadIdx.x; j < Bb * Tt * Pp; j += 256) {
    int b = j / (Tt * Pp), i = j % (Tt * Pp);
    float v = x[(size_t)(b * Cc + c) * (Tt * Pp) + i];
    s += v; q += v * v;
  }
  ssum[threadIdx.x] = s; ssq[threadIdx.x] = q;
  __syncthreads();
  for (int st = 128; st > 0; st >>= 1) {
    if (threadIdx.x < st) {
      ssum[threadIdx.x] += ssum[threadIdx.x + st];
      ssq[threadIdx.x]  += ssq[threadIdx.x + st];
    }
    __syncthreads();
  }
  if (threadIdx.x == 0) {
    float m = ssum[0] / (float)(Bb * Tt * Pp);
    float v = ssq[0] / (float)(Bb * Tt * Pp) - m * m;
    mean[c] = m;
    rstd[c] = rsqrtf(v + EPS);
  }
}

// ================= kernel 2: weight prep (bf16 + Wo transpose + bo mean) =================
__global__ __launch_bounds__(256) void wprep_kernel(
    const float* __restrict__ Wq, const float* __restrict__ Wk, const float* __restrict__ Wv,
    const float* __restrict__ Wo, const float* __restrict__ bo,
    unsigned short* Wqh, unsigned short* Wkh, unsigned short* Wvh,
    unsigned short* Woth, float* bo_mean) {
  int which = blockIdx.y;
  int i = blockIdx.x * 256 + threadIdx.x;
  if (which < 3) {
    const float* src = which == 0 ? Wq : (which == 1 ? Wk : Wv);
    unsigned short* dst = which == 0 ? Wqh : (which == 1 ? Wkh : Wvh);
    if (i < NH * Ee * Cc) dst[i] = f2bf(src[i]);
  } else if (which == 3) {
    if (i < Cc * NH * Ee) {                       // Woth[c][(n,e)] = Wo[n][c][e]
      int c = i >> 9, j = i & 511, n = j >> 6, e = j & 63;
      Woth[i] = f2bf(Wo[((size_t)n * Cc + c) * Ee + e]);
    }
  } else {
    if (i < Cc) {
      float s = 0.f;
      for (int n = 0; n < NH; ++n) s += bo[n * Cc + i];
      bo_mean[i] = s * (1.f / NH);
    }
  }
}

// ================= kernel 3: normalize -> xh bf16 + d_out init =================
__global__ __launch_bounds__(256) void xprep_kernel(
    const float* __restrict__ x, const float* __restrict__ gamma, const float* __restrict__ beta,
    const float* __restrict__ mean, const float* __restrict__ rstd, const float* __restrict__ bo_mean,
    unsigned short* xh, float* out) {
  size_t idx = (size_t)blockIdx.x * 256 + threadIdx.x;   // B*T*64*C = 8388608
  if (idx >= (size_t)Bb * Tt * P64 * Cc) return;
  int c  = (int)(idx & 511);
  int p  = (int)((idx >> 9) & 63);
  int t  = (int)((idx >> 15) & 31);
  int b  = (int)(idx >> 20);
  unsigned short h = 0;
  if (p < Pp) {
    size_t xi = ((size_t)(b * Cc + c) * Tt + t) * Pp + p;
    float v  = x[xi];
    float xb = gamma[c] * (v - mean[c]) * rstd[c] + beta[c];
    h = f2bf(xb);
    out[xi] = xb + bo_mean[c];      // exact f32 residual + mean head-bias
  }
  xh[idx] = h;                      // padded p rows = 0
}

// ================= kernel 4: q/k/v projections (WMMA, x slice staged in LDS) =================
// One block per (bt, which): all 8 heads share the LDS-resident activation slice.
__global__ __launch_bounds__(256) void proj_kernel(
    const unsigned short* __restrict__ Wqh, const unsigned short* __restrict__ Wkh,
    const unsigned short* __restrict__ Wvh,
    const float* __restrict__ bq, const float* __restrict__ bk, const float* __restrict__ bv,
    const unsigned short* __restrict__ xh,
    unsigned short* qT, unsigned short* kT, unsigned short* vE) {
  __shared__ unsigned short xs[P64 * Cc];                  // 64 KB: B^T slice [p64][C]
  int bt = blockIdx.x, which = blockIdx.y;
  int b = bt / Tt, t = bt % Tt;
  int tid = threadIdx.x, lane = tid & 31, wave = tid >> 5;
  const unsigned short* Wh = (which == 0 ? Wqh : (which == 1 ? Wkh : Wvh));
  const float* bias = which == 0 ? bq : (which == 1 ? bk : bv);

  const unsigned short* xg = xh + (size_t)bt * P64 * Cc;
  for (int i = tid; i < (P64 * Cc) / 8; i += 256)
    *(v8us*)(xs + i * 8) = *(const v8us*)(xg + (size_t)i * 8);
  __syncthreads();

  // 8 heads x 16 tiles = 128 tiles over 8 waves
  for (int tt = wave; tt < NH * 16; tt += 8) {
    int n  = tt >> 4;
    int mi = (tt >> 2) & 3, ni = tt & 3;                   // mi: e-tile, ni: p-tile
    const unsigned short* Wn = Wh + (size_t)n * Ee * Cc;
    v8f acc = {};
    for (int kc = 0; kc < Cc / 32; ++kc) {
      v16bf a = load_fragA(Wn, Cc, mi * 16, kc * 32, lane);
      v16bf bfr = load_fragB(xs, Cc, ni * 16, kc * 32, lane);
      acc = wmma_bf16(a, bfr, acc);
    }
    int off = (lane >> 4) * 8;
    int p = ni * 16 + (lane & 15);
    size_t slice = ((size_t)(b * NH + n) * Tt + t);
    if (which < 2) {                                       // qT/kT [p][e], contiguous e
      unsigned short* dst = (which == 0 ? qT : kT);
      size_t base = (slice * P64 + p) * Ee + mi * 16 + off;
      v8us pk;
      for (int r = 0; r < 8; ++r) {
        int e = mi * 16 + off + r;
        pk[r] = f2bf(acc[r] + bias[n * Ee + e]);
      }
      *(v8us*)(dst + base) = pk;
    } else {                                               // vE [e][p]
      size_t base = slice * Ee * P64;
      for (int r = 0; r < 8; ++r) {
        int e = mi * 16 + off + r;
        vE[base + (size_t)e * P64 + p] = f2bf(acc[r] + bias[n * Ee + e]);
      }
    }
  }
}

// ================= kernel 5: windowed attention per (b,t,n) =================
__global__ __launch_bounds__(256) void attn_kernel(
    const unsigned short* __restrict__ qT, const unsigned short* __restrict__ kT,
    const unsigned short* __restrict__ vE,
    const float* __restrict__ bk, const float* __restrict__ bv,
    unsigned short* yT) {
  extern __shared__ char smem[];
  unsigned short* k_lds  = (unsigned short*)smem;                          // [352][64]
  unsigned short* v_lds  = (unsigned short*)(smem + SM_K);                 // [64][352]
  unsigned short* q_lds  = (unsigned short*)(smem + SM_K + SM_V);          // [64][64]
  float*          alphaf = (float*)(smem + SM_K + SM_V + SM_Q);            // [64][352]
  unsigned short* alphah = (unsigned short*)(smem + SM_K + SM_V + SM_Q + SM_AF); // [64][352]

  int bt = blockIdx.x, n = blockIdx.y;
  int b = bt / Tt, t = bt % Tt;
  int tid = threadIdx.x, lane = tid & 31, wave = tid >> 5;
  size_t slice = (size_t)(b * NH + n) * Tt;

  // ---- stage q[p][e] ----
  const unsigned short* qsrc = qT + (slice + t) * P64 * Ee;
  for (int i = tid; i < (64 * 64) / 8; i += 256)
    *(v8us*)(q_lds + i * 8) = *(const v8us*)(qsrc + (size_t)i * 8);

  // ---- stage k[l][e]; out-of-window time slices get bias (matches zero-padded xb) ----
  for (int i = tid; i < L352 * 8; i += 256) {
    int l = i >> 3, c8 = (i & 7) * 8;
    v8us val;
    if (l < Ll) {
      int kk = l / Pp, p = l - kk * Pp;
      int tp = t + kk - (KW / 2);
      if (tp >= 0 && tp < Tt) {
        val = *(const v8us*)(kT + ((slice + tp) * P64 + p) * Ee + c8);
      } else {
        for (int r = 0; r < 8; ++r) val[r] = f2bf(bk[n * Ee + c8 + r]);
      }
    } else {
      for (int r = 0; r < 8; ++r) val[r] = 0;
    }
    *(v8us*)(k_lds + (size_t)l * 64 + c8) = val;
  }

  // ---- stage v[e][l] ----
  for (int i = tid; i < 64 * L352; i += 256) {
    int e = i / L352, l = i - e * L352;
    unsigned short val = 0;
    if (l < Ll) {
      int kk = l / Pp, p = l - kk * Pp;
      int tp = t + kk - (KW / 2);
      val = (tp >= 0 && tp < Tt) ? vE[((slice + tp) * Ee + e) * P64 + p]
                                 : f2bf(bv[n * Ee + e]);
    }
    v_lds[(size_t)e * L352 + l] = val;
  }
  __syncthreads();

  // ---- logits: alpha[p][l] = q^T k ; 4 x 22 tiles, K=E=64 ----
  for (int tt = wave; tt < 4 * 22; tt += 8) {
    int mi = tt / 22, ni = tt % 22;
    v8f acc = {};
    for (int kc = 0; kc < 2; ++kc) {
      v16bf a = load_fragA(q_lds, Ee, mi * 16, kc * 32, lane);
      v16bf bfr = load_fragB(k_lds, Ee, ni * 16, kc * 32, lane);
      acc = wmma_bf16(a, bfr, acc);
    }
    int off = (lane >> 4) * 8, lcol = ni * 16 + (lane & 15);
    for (int r = 0; r < 8; ++r)
      alphaf[(size_t)(mi * 16 + off + r) * L352 + lcol] = acc[r];
  }
  __syncthreads();

  // ---- parallel row softmax: 4 lanes per row, shfl_xor reductions ----
  {
    int row = tid >> 2, q4 = tid & 3;                // rows 0..63, quarter chunks of 88
    float* rowp = alphaf + (size_t)row * L352;
    unsigned short* ah = alphah + (size_t)row * L352;
    int l0 = q4 * (L352 / 4);
    if (row < Pp) {
      int l1 = min(l0 + L352 / 4, Ll);
      float mx = -1e30f;
      for (int l = l0; l < l1; ++l) mx = fmaxf(mx, rowp[l]);
      for (int m = 1; m < 4; m <<= 1) mx = fmaxf(mx, __shfl_xor(mx, m, 32));
      float s = 0.f;
      for (int l = l0; l < l1; ++l) { float e = __expf(rowp[l] - mx); rowp[l] = e; s += e; }
      for (int m = 1; m < 4; m <<= 1) s += __shfl_xor(s, m, 32);
      float inv = 1.f / s;
      for (int l = l0; l < l1; ++l) ah[l] = f2bf(rowp[l] * inv);
      if (q4 == 3)
        for (int l = Ll; l < L352; ++l) ah[l] = 0;   // pad columns
    } else {
      for (int l = l0; l < l0 + L352 / 4; ++l) ah[l] = 0;  // padded p rows -> zero weights
    }
  }
  __syncthreads();

  // ---- y^T[p][e] = alpha (PxL) @ v^T (LxE) ; 4 x 4 tiles, K=352 ----
  for (int tt = wave * 2; tt < wave * 2 + 2; ++tt) {
    int mi = tt >> 2, ni = tt & 3;
    v8f acc = {};
    for (int kc = 0; kc < L352 / 32; ++kc) {
      v16bf a = load_fragA(alphah, L352, mi * 16, kc * 32, lane);
      v16bf bfr = load_fragB(v_lds, L352, ni * 16, kc * 32, lane);
      acc = wmma_bf16(a, bfr, acc);
    }
    int off = (lane >> 4) * 8, e = ni * 16 + (lane & 15);
    for (int r = 0; r < 8; ++r) {
      int p = mi * 16 + off + r;
      yT[(((size_t)bt * P64 + p) * NH + n) * Ee + e] = f2bf(acc[r]);
    }
  }
}

// ================= kernel 6: fused output projection over all heads =================
__global__ __launch_bounds__(256) void outproj_kernel(
    const unsigned short* __restrict__ Woth, const unsigned short* __restrict__ yT,
    float* out) {
  __shared__ unsigned short yb[64 * 512];                    // 64 KB: y^T slice [p][(n,e)]
  int bt = blockIdx.x;
  int b = bt / Tt, t = bt % Tt;
  int tid = threadIdx.x, lane = tid & 31, wave = tid >> 5;
  const unsigned short* ys = yT + (size_t)bt * P64 * (NH * Ee);
  for (int i = tid; i < (64 * 512) / 8; i += 256)
    *(v8us*)(yb + i * 8) = *(const v8us*)(ys + (size_t)i * 8);
  __syncthreads();

  // GEMM: M=C=512 (32 tiles), N=64 (4 tiles), K=NH*E=512 (16 WMMA)
  for (int tt = wave; tt < 32 * 4; tt += 8) {
    int mi = tt >> 2, ni = tt & 3;
    v8f acc = {};
    for (int kc = 0; kc < (NH * Ee) / 32; ++kc) {
      v16bf a = load_fragA(Woth, NH * Ee, mi * 16, kc * 32, lane);
      v16bf bfr = load_fragB(yb, NH * Ee, ni * 16, kc * 32, lane);
      acc = wmma_bf16(a, bfr, acc);
    }
    int off = (lane >> 4) * 8;
    int p = ni * 16 + (lane & 15);
    if (p < Pp) {
      for (int r = 0; r < 8; ++r) {
        int c = mi * 16 + off + r;
        size_t oi = ((size_t)(b * Cc + c) * Tt + t) * Pp + p;
        out[oi] += acc[r] * (1.f / NH);     // deterministic: unique (b,t) per block
      }
    }
  }
}

// ================= host launcher =================
extern "C" void kernel_launch(void* const* d_in, const int* in_sizes, int n_in,
                              void* d_out, int out_size, void* d_ws, size_t ws_size,
                              hipStream_t stream) {
  const float* x     = (const float*)d_in[0];
  const float* gamma = (const float*)d_in[1];
  const float* beta  = (const float*)d_in[2];
  const float* Wq    = (const float*)d_in[3];
  const float* bq    = (const float*)d_in[4];
  const float* Wk    = (const float*)d_in[5];
  const float* bk    = (const float*)d_in[6];
  const float* Wv    = (const float*)d_in[7];
  const float* bv    = (const float*)d_in[8];
  const float* Wo    = (const float*)d_in[9];
  const float* bo    = (const float*)d_in[10];
  float* out = (float*)d_out;

  char* w = (char*)d_ws;
  float* mean    = (float*)(w + OFF_MEAN);
  float* rstd    = (float*)(w + OFF_RSTD);
  float* bo_mean = (float*)(w + OFF_BOM);
  unsigned short* Wqh  = (unsigned short*)(w + OFF_WQH);
  unsigned short* Wkh  = (unsigned short*)(w + OFF_WKH);
  unsigned short* Wvh  = (unsigned short*)(w + OFF_WVH);
  unsigned short* Woth = (unsigned short*)(w + OFF_WOTH);
  unsigned short* xh   = (unsigned short*)(w + OFF_XH);
  unsigned short* qT   = (unsigned short*)(w + OFF_QT);
  unsigned short* kT   = (unsigned short*)(w + OFF_KT);
  unsigned short* vE   = (unsigned short*)(w + OFF_VE);
  unsigned short* yT   = (unsigned short*)(w + OFF_YT);

  // allow 228 KB dynamic LDS for the attention kernel (WGP supports up to 320 KB)
  (void)hipFuncSetAttribute((const void*)attn_kernel,
                            hipFuncAttributeMaxDynamicSharedMemorySize, (int)SMEM_ATTN);

  bnstats_kernel<<<Cc, 256, 0, stream>>>(x, mean, rstd);
  wprep_kernel<<<dim3((NH * Ee * Cc) / 256, 5), 256, 0, stream>>>(
      Wq, Wk, Wv, Wo, bo, Wqh, Wkh, Wvh, Woth, bo_mean);
  xprep_kernel<<<(int)(((size_t)Bb * Tt * P64 * Cc) / 256), 256, 0, stream>>>(
      x, gamma, beta, mean, rstd, bo_mean, xh, out);
  proj_kernel<<<dim3(Bb * Tt, 3), 256, 0, stream>>>(
      Wqh, Wkh, Wvh, bq, bk, bv, xh, qT, kT, vE);
  attn_kernel<<<dim3(Bb * Tt, NH), 256, SMEM_ATTN, stream>>>(qT, kT, vE, bk, bv, yT);
  outproj_kernel<<<Bb * Tt, 256, 0, stream>>>(Woth, yT, out);
}
